// DiagnosticRNN_43387759624469
// MI455X (gfx1250) — matrix operand order repaired and checked by reading, not compiled, not run on hardware
//
#include <hip/hip_runtime.h>

// Problem constants
#define B_TOT 16384
#define S_LEN 64
#define V_DIM 25
#define E_DIM 64
#define H_DIM 64
#define G_DIM 256   // 4*H
#define C_DIM 3
#define KPAD  104   // padded K stride (bf16 elems): 208 B = 52 dwords -> conflict-free b128 frags

typedef __attribute__((ext_vector_type(16))) __bf16 v16bf;
typedef __attribute__((ext_vector_type(8)))  float  v8f;

union Frag {
  v16bf v;
  uint4 q[2];
  __bf16 h[16];
};

union BfPair {
  __bf16 h[2];
  unsigned u;
};

__device__ __forceinline__ float fast_sigmoid(float x) {
  return 1.0f / (1.0f + __expf(-x));
}
__device__ __forceinline__ float fast_tanh(float x) {
  return 1.0f - 2.0f / (__expf(2.0f * x) + 1.0f);
}

__global__ __launch_bounds__(128) void lstm_fused_kernel(
    const float* __restrict__ msgs,   // [B,S,V]
    const float* __restrict__ emb,    // [V,E]
    const float* __restrict__ W_ih,   // [4H,E]
    const float* __restrict__ W_hh,   // [4H,H]
    const float* __restrict__ b_ih,   // [4H]
    const float* __restrict__ b_hh,   // [4H]
    const float* __restrict__ fc_w,   // [C,H]
    const float* __restrict__ fc_b,   // [C]
    float* __restrict__ out)          // [B,C]
{
  // W_cat in LDS, [n][k] bf16: k<64 -> W_hh^T, 64..88 -> W_comb = embedding@W_ih^T, rest 0
  __shared__ __bf16 Wlds[G_DIM * KPAD];        // 53248 B
  __shared__ __bf16 Xlds[4 * 16 * KPAD];       // per-wave h staging, [row][k] bf16

  const int tid  = threadIdx.x;
  const int wave = tid >> 5;
  const int lane = tid & 31;
  const int lm   = lane & 15;   // column-within-tile (B/C layouts) / row M (A layout)
  const int sub  = lane >> 4;   // lane half

  // ================= Prologue: build W_cat in LDS =================
  // (a) W_hh^T part (cols 0..63): packed f32 -> 2x bf16 copy (native cvt, pair stores)
  for (int n = tid; n < G_DIM; n += 128) {
    const float* whh_row = W_hh + n * H_DIM;
    unsigned* dst32 = (unsigned*)&Wlds[n * KPAD];
    #pragma unroll 4
    for (int k2 = 0; k2 < H_DIM / 2; ++k2) {
      BfPair p;
      p.h[0] = (__bf16)whh_row[2 * k2];
      p.h[1] = (__bf16)whh_row[2 * k2 + 1];
      dst32[k2] = p.u;
    }
    // zero the tail pad columns (89..103); cols 64..88 written by WMMA below
    for (int k = H_DIM + V_DIM; k < KPAD; ++k) Wlds[n * KPAD + k] = (__bf16)0.f;
  }

  // (b) W_comb part (cols 64..88) via WMMA: [25x64] @ [64x256] -> [25x256]
  //     wave w owns n-tiles w*4 .. w*4+3; M padded to 32 (2 tiles)
  {
    // dummy sink for out-of-range scatter lanes: this wave's own Xlds region,
    // which is zero-filled by this wave right after (DS ops in-order per wave).
    __bf16* dump = &Xlds[wave * 16 * KPAD + lane];

    Frag aE[2][2];   // [mt][kstep] embedding A-fragments
    #pragma unroll
    for (int mt = 0; mt < 2; ++mt) {
      int row = mt * 16 + lm;
      int rc  = row < V_DIM ? row : 0;
      const float* er = emb + rc * E_DIM;
      #pragma unroll
      for (int ks = 0; ks < 2; ++ks) {
        #pragma unroll
        for (int i = 0; i < 8; ++i) {
          float x0 = er[ks * 32 + 8 * sub + i];
          float x1 = er[ks * 32 + 16 + 8 * sub + i];
          if (row >= V_DIM) { x0 = 0.f; x1 = 0.f; }
          aE[mt][ks].h[i]     = (__bf16)x0;
          aE[mt][ks].h[8 + i] = (__bf16)x1;
        }
      }
    }
    #pragma unroll
    for (int nti = 0; nti < 4; ++nti) {
      int n = (wave * 4 + nti) * 16 + lm;
      Frag bW[2];
      #pragma unroll
      for (int ks = 0; ks < 2; ++ks)
        #pragma unroll
        for (int i = 0; i < 16; ++i)
          bW[ks].h[i] = (__bf16)W_ih[n * E_DIM + ks * 32 + 8 * sub + (i & 7) + ((i >> 3) << 4)];
      #pragma unroll
      for (int mt = 0; mt < 2; ++mt) {
        v8f c = {0.f, 0.f, 0.f, 0.f, 0.f, 0.f, 0.f, 0.f};
        c = __builtin_amdgcn_wmma_f32_16x16x32_bf16(false, aE[mt][0].v, false, bW[0].v, (short)0, c, false, false);
        c = __builtin_amdgcn_wmma_f32_16x16x32_bf16(false, aE[mt][1].v, false, bW[1].v, (short)0, c, false, false);
        #pragma unroll
        for (int r = 0; r < 8; ++r) {
          int v = mt * 16 + r + 8 * sub;            // vocab row
          // branch-free scatter: clamp OOB lanes to the dump slot
          __bf16* dst = (v < V_DIM) ? &Wlds[n * KPAD + H_DIM + v] : dump;
          *dst = (__bf16)c[r];
        }
      }
    }
  }

  // zero this wave's h-staging tile (h0 = 0); also scrubs the dump slot
  {
    __bf16* xr = &Xlds[wave * 16 * KPAD];
    for (int i = lane; i < 16 * KPAD; i += 32) xr[i] = (__bf16)0.f;
  }
  __syncthreads();

  // per-lane bias (added once per step, matching reference b_ih + b_hh)
  float bias_n[16];
  #pragma unroll
  for (int nt = 0; nt < 16; ++nt) {
    int n = nt * 16 + lm;
    bias_n[nt] = b_ih[n] + b_hh[n];
  }

  const int rowBase = blockIdx.x * 64 + wave * 16;
  const float* mrow = msgs + (size_t)(rowBase + lm) * (S_LEN * V_DIM); // A-layout: lane row = lm

  float cS[4][8], hS[4][8];
  #pragma unroll
  for (int j = 0; j < 4; ++j)
    #pragma unroll
    for (int r = 0; r < 8; ++r) { cS[j][r] = 0.f; hS[j][r] = 0.f; }

  __bf16* xrow = &Xlds[(wave * 16 + lm) * KPAD];

  // ================= Main recurrence =================
  #pragma unroll 1
  for (int t = 0; t < S_LEN; ++t) {
    // --- A fragments: K 0..63 = h (from LDS staging), K 64..95 = message slice (global) ---
    Frag a0, a1, a2;
    a0.q[0] = *(const uint4*)&xrow[ 0 + sub * 8];
    a0.q[1] = *(const uint4*)&xrow[16 + sub * 8];
    a1.q[0] = *(const uint4*)&xrow[32 + sub * 8];
    a1.q[1] = *(const uint4*)&xrow[48 + sub * 8];
    const float* mt = mrow + t * V_DIM;
    #pragma unroll
    for (int i = 0; i < 8; ++i) {
      int v0 = sub * 8 + i;            // 0..15 (always valid)
      int v1 = 16 + sub * 8 + i;       // 16..31
      int c1 = v1 < V_DIM ? v1 : 0;
      float x0 = mt[v0];
      float x1 = mt[c1]; if (v1 >= V_DIM) x1 = 0.f;
      a2.h[i]     = (__bf16)x0;
      a2.h[8 + i] = (__bf16)x1;
    }
    // keep the HBM-bound message stream ahead of the recurrence
    {
      int tn = t + 1 < S_LEN ? t + 1 : S_LEN - 1;
      __builtin_prefetch(mrow + tn * V_DIM, 0, 3);
    }

    // --- per h-slice j: 4 gate tiles (i,f,g,o) x 3 chained WMMAs, then the LSTM cell ---
    #pragma unroll
    for (int j = 0; j < 4; ++j) {
      v8f g[4];
      #pragma unroll
      for (int q = 0; q < 4; ++q) {
        const int nt = q * 4 + j;                 // i: nt 0..3, f: 4..7, g: 8..11, o: 12..15
        float b = bias_n[nt];
        v8f a = {b, b, b, b, b, b, b, b};
        const __bf16* wp = &Wlds[(nt * 16 + lm) * KPAD + sub * 8];
        Frag b0, b1, b2;
        b0.q[0] = *(const uint4*)&wp[ 0]; b0.q[1] = *(const uint4*)&wp[16];
        b1.q[0] = *(const uint4*)&wp[32]; b1.q[1] = *(const uint4*)&wp[48];
        b2.q[0] = *(const uint4*)&wp[64]; b2.q[1] = *(const uint4*)&wp[80];
        a = __builtin_amdgcn_wmma_f32_16x16x32_bf16(false, a0.v, false, b0.v, (short)0, a, false, false);
        a = __builtin_amdgcn_wmma_f32_16x16x32_bf16(false, a1.v, false, b1.v, (short)0, a, false, false);
        a = __builtin_amdgcn_wmma_f32_16x16x32_bf16(false, a2.v, false, b2.v, (short)0, a, false, false);
        g[q] = a;
      }
      #pragma unroll
      for (int r = 0; r < 8; ++r) {
        float iv = fast_sigmoid(g[0][r]);
        float fv = fast_sigmoid(g[1][r]);
        float gv = fast_tanh   (g[2][r]);
        float ov = fast_sigmoid(g[3][r]);
        float cn = fv * cS[j][r] + iv * gv;
        float hn = ov * fast_tanh(cn);
        cS[j][r] = cn;
        hS[j][r] = hn;
        // restage h for next step's A fragment: row M = r + 8*sub, col K = 16j + lm
        Xlds[(wave * 16 + r + 8 * sub) * KPAD + j * 16 + lm] = (__bf16)hn;
      }
    }
    asm volatile("s_wait_dscnt 0" ::: "memory");  // staging visible before next-step loads
  }

  // ================= fc tail: out = h @ fc_w^T + fc_b =================
  __syncthreads();                                   // done with Wlds
  float* hbuf = (float*)Wlds + wave * (16 * H_DIM);  // 4 KB per wave
  #pragma unroll
  for (int j = 0; j < 4; ++j)
    #pragma unroll
    for (int r = 0; r < 8; ++r)
      hbuf[(r + 8 * sub) * H_DIM + j * 16 + lm] = hS[j][r];
  asm volatile("s_wait_dscnt 0" ::: "memory");

  if (lane < 16) {
    const float* hr = hbuf + lane * H_DIM;
    #pragma unroll
    for (int cc = 0; cc < C_DIM; ++cc) {
      float s = fc_b[cc];
      const float* wr = fc_w + cc * H_DIM;
      #pragma unroll 8
      for (int k = 0; k < H_DIM; ++k) s += hr[k] * wr[k];
      out[(size_t)(rowBase + lane) * C_DIM + cc] = s;
    }
  }
}

extern "C" void kernel_launch(void* const* d_in, const int* in_sizes, int n_in,
                              void* d_out, int out_size, void* d_ws, size_t ws_size,
                              hipStream_t stream) {
  (void)in_sizes; (void)n_in; (void)d_ws; (void)ws_size; (void)out_size;
  const float* msgs = (const float*)d_in[0];
  const float* emb  = (const float*)d_in[1];
  const float* W_ih = (const float*)d_in[2];
  const float* W_hh = (const float*)d_in[3];
  const float* b_ih = (const float*)d_in[4];
  const float* b_hh = (const float*)d_in[5];
  const float* fc_w = (const float*)d_in[6];
  const float* fc_b = (const float*)d_in[7];
  float* out = (float*)d_out;

  dim3 grid(B_TOT / 64);   // 256 WGs, 64 batch rows each (4 waves x 16 rows)
  dim3 block(128);
  lstm_fused_kernel<<<grid, block, 0, stream>>>(msgs, emb, W_ih, W_hh, b_ih, b_hh,
                                                fc_w, fc_b, out);
}
